// CoAttentionLayer3_67886253080717
// MI455X (gfx1250) — matrix operand order repaired and checked by reading, not compiled, not run on hardware
//
#include <hip/hip_runtime.h>

typedef __attribute__((ext_vector_type(16))) _Float16 v16h;
typedef __attribute__((ext_vector_type(8)))  _Float16 v8h;
typedef __attribute__((ext_vector_type(8)))  float    v8f;

#define NTOK   16
#define DIM    512
#define HEADS  16
#define DHEAD  64
#define INNER  1024
#define THREADS 512
// xh 32K | d1h 32K | d1t 32K | d2h 32K | Ah 8K | o1 4K | red 8B
#define OFF_XH   0
#define OFF_D1H  32768
#define OFF_D1T  65536
#define OFF_D2H  98304
#define OFF_AH   131072
#define OFF_O1   139264
#define OFF_RED  143360
#define SMEM_BYTES 143376

static __device__ __forceinline__ v8f wmma16(v16h a, v16h b, v8f c) {
  return __builtin_amdgcn_wmma_f32_16x16x32_f16(false, a, false, b, (short)0, c,
                                                false, false);
}

// A-operand (16-bit 16x32): lane-half contiguous K (k = 16h + s) in two 8-half runs.
static __device__ __forceinline__ v16h load_a(const _Float16* p0, const _Float16* p1) {
  v8h lo = *(const v8h*)p0;
  v8h hi = *(const v8h*)p1;
  return __builtin_shufflevector(lo, hi, 0,1,2,3,4,5,6,7,8,9,10,11,12,13,14,15);
}

// B-operand (16-bit 32x16): 16 consecutive K values per lane, from f32 memory.
static __device__ __forceinline__ v16h load_b_f32(const float* p) {
  float4 a = ((const float4*)p)[0];
  float4 b = ((const float4*)p)[1];
  float4 c = ((const float4*)p)[2];
  float4 d = ((const float4*)p)[3];
  v16h r;
  r[0]=(_Float16)a.x;  r[1]=(_Float16)a.y;  r[2]=(_Float16)a.z;  r[3]=(_Float16)a.w;
  r[4]=(_Float16)b.x;  r[5]=(_Float16)b.y;  r[6]=(_Float16)b.z;  r[7]=(_Float16)b.w;
  r[8]=(_Float16)c.x;  r[9]=(_Float16)c.y;  r[10]=(_Float16)c.z; r[11]=(_Float16)c.w;
  r[12]=(_Float16)d.x; r[13]=(_Float16)d.y; r[14]=(_Float16)d.z; r[15]=(_Float16)d.w;
  return r;
}

__global__ __launch_bounds__(THREADS)
void coattn_fused_kernel(const float* __restrict__ drug1,
                         const float* __restrict__ drug2,
                         const float* __restrict__ ln_w,
                         const float* __restrict__ ln_b,
                         const float* __restrict__ Wd,
                         const float* __restrict__ Wout,
                         const float* __restrict__ bout,
                         float* __restrict__ out) {
  extern __shared__ char smem[];
  _Float16* xh  = (_Float16*)(smem + OFF_XH);   // [32][512]  f16 (x1 rows 0-15, x2 rows 16-31)
  _Float16* d1h = (_Float16*)(smem + OFF_D1H);  // [16][1024] f16
  _Float16* d1t = (_Float16*)(smem + OFF_D1T);  // [1024][16] f16 (transposed d1)
  _Float16* d2h = (_Float16*)(smem + OFF_D2H);  // [16][1024] f16
  _Float16* Ah  = (_Float16*)(smem + OFF_AH);   // [16][256]  f16 softmax probs
  float*    o1  = (float*)(smem + OFF_O1);      // [1024] f32
  float*    red = (float*)(smem + OFF_RED);     // [2]

  const int b     = blockIdx.x;
  const int tid   = threadIdx.x;
  const int lane  = tid & 31;
  const int wave  = tid >> 5;
  const int l15   = lane & 15;
  const int lhalf = lane >> 4;

  if (tid == 0) { red[0] = 0.f; red[1] = 0.f; }

  // ---- Phase 1: LayerNorm. Wave w normalizes token w of drug1 and drug2. ----
#pragma unroll
  for (int s = 0; s < 2; ++s) {
    const float* src = (s == 0 ? drug1 : drug2) + ((size_t)b * NTOK + wave) * DIM;
    float fv[16];
    float sum = 0.f, sq = 0.f;
#pragma unroll
    for (int i = 0; i < 4; ++i) {
      float4 t = ((const float4*)src)[lane + 32 * i];
      fv[4*i+0]=t.x; fv[4*i+1]=t.y; fv[4*i+2]=t.z; fv[4*i+3]=t.w;
      sum += t.x + t.y + t.z + t.w;
      sq  += t.x*t.x + t.y*t.y + t.z*t.z + t.w*t.w;
    }
#pragma unroll
    for (int m = 16; m >= 1; m >>= 1) {
      sum += __shfl_xor(sum, m, 32);
      sq  += __shfl_xor(sq,  m, 32);
    }
    const float mu   = sum * (1.f / DIM);
    const float var  = sq * (1.f / DIM) - mu * mu;
    const float rstd = rsqrtf(var + 1e-5f);
    _Float16* dstrow = xh + (size_t)(s * NTOK + wave) * DIM;
#pragma unroll
    for (int i = 0; i < 4; ++i)
#pragma unroll
      for (int c = 0; c < 4; ++c) {
        const int k = (lane + 32 * i) * 4 + c;
        dstrow[k] = (_Float16)((fv[4*i+c] - mu) * rstd * ln_w[k] + ln_b[k]);
      }
  }
  __syncthreads();

  // ---- Phase 2: d1 = x1@Wd^T, d2 = x2@Wd^T (B tile shared between both). ----
  {
    const _Float16* a1base = xh + (size_t)l15 * DIM;
    const _Float16* a2base = a1base + (size_t)NTOK * DIM;
    const int kh   = lhalf * 16;  // A K-run start within 32-chunk
    const int kb16 = lhalf * 16;  // B K-run start within 32-chunk
#pragma unroll
    for (int tt = 0; tt < 4; ++tt) {
      const int cb = (wave + tt * 16) * 16;   // output column base
      const float* bbase = Wd + (size_t)(cb + l15) * DIM;
      v8f c1 = {}; v8f c2 = {};
      for (int ks = 0; ks < 16; ++ks) {
        const int kb = ks * 32;
        if (ks < 15) __builtin_prefetch(bbase + kb + 32 + kb16, 0, 1);
        v16h a1 = load_a(a1base + kb + kh, a1base + kb + 8 + kh);
        v16h a2 = load_a(a2base + kb + kh, a2base + kb + 8 + kh);
        v16h bb = load_b_f32(bbase + kb + kb16);
        c1 = wmma16(a1, bb, c1);
        c2 = wmma16(a2, bb, c2);
      }
      const int j = cb + l15;
#pragma unroll
      for (int r = 0; r < 8; ++r) {
        const int m = r + lhalf * 8;            // token row
        const _Float16 h1 = (_Float16)c1[r];
        d1h[(size_t)m * INNER + j] = h1;
        d1t[(size_t)j * NTOK + m]  = h1;
        d2h[(size_t)m * INNER + j] = (_Float16)c2[r];
      }
    }
  }
  __syncthreads();

  // ---- Phase 3: per-head scores + joint softmax + out1b. Wave w == head h. ----
  {
    const int h  = wave;
    const int kh = lhalf * 16;
    const _Float16* aab = d1h + (size_t)l15 * INNER + h * DHEAD; // A row n = l15
    const _Float16* bbb = d2h + (size_t)l15 * INNER + h * DHEAD; // B col m = l15
    v8f acc = {};
#pragma unroll
    for (int ks = 0; ks < 2; ++ks) {
      const int kb = ks * 32;
      v16h aa = load_a(aab + kb + kh, aab + kb + 8 + kh);
      v16h bb = *(const v16h*)(bbb + kb + lhalf * 16);
      acc = wmma16(aa, bb, acc);
    }
    // joint softmax over all 256 scores of this head (entirely in one wave)
    float e[8]; float mx = -3.4e38f;
#pragma unroll
    for (int r = 0; r < 8; ++r) { e[r] = acc[r] * 0.125f; mx = fmaxf(mx, e[r]); }
#pragma unroll
    for (int m = 16; m >= 1; m >>= 1) mx = fmaxf(mx, __shfl_xor(mx, m, 32));
    float ssum = 0.f;
#pragma unroll
    for (int r = 0; r < 8; ++r) { e[r] = __expf(e[r] - mx); ssum += e[r]; }
#pragma unroll
    for (int m = 16; m >= 1; m >>= 1) ssum += __shfl_xor(ssum, m, 32);
    const float inv = 1.f / ssum;
    float S = 0.f;  // column sums S[h, k=l15] = sum_q A[h,q,k]
#pragma unroll
    for (int r = 0; r < 8; ++r) {
      const float p = e[r] * inv; S += p;
      const int q = r + lhalf * 8;
      Ah[(size_t)h * 256 + q * 16 + l15] = (_Float16)p;
    }
    S += __shfl_xor(S, 16, 32);
    // out1b[h,d] = sum_k S[k] * d2[h,k,d]; lane handles d = lane, lane+32
    float a0 = 0.f, a1 = 0.f;
    const _Float16* db = d2h + h * DHEAD;
#pragma unroll
    for (int k = 0; k < 16; ++k) {
      const float sk = __shfl(S, k, 32);
      a0 += sk * (float)db[(size_t)k * INNER + lane];
      a1 += sk * (float)db[(size_t)k * INNER + lane + 32];
    }
    o1[h * DHEAD + lane]      = a0;
    o1[h * DHEAD + lane + 32] = a1;
  }
  __syncthreads();

  // ---- Phase 4: out1a = A_flat(16x256) @ d1_flat(256x64); waves 0-3, K=256. ----
  if (wave < 4) {
    const int kh   = lhalf * 16;
    const int dabs = wave * 16 + l15;           // output column d
    const _Float16* ab = Ah + (size_t)l15 * 256; // A row i = l15
    v8f acc = {};
#pragma unroll
    for (int ks = 0; ks < 8; ++ks) {
      const int kb = ks * 32;
      v16h aa = load_a(ab + kb + kh, ab + kb + 8 + kh);
      const int q = (kb + lhalf * 16) >> 4;     // qk run start /16 = head-of-d1 index
      v16h bb = *(const v16h*)(d1t + (size_t)(q * DHEAD + dabs) * NTOK);
      acc = wmma16(aa, bb, acc);
    }
#pragma unroll
    for (int r = 0; r < 8; ++r) {
      const int i = r + lhalf * 8;              // head row
      o1[i * DHEAD + dabs] += acc[r];
    }
  }
  __syncthreads();

  // ---- Phase 5: out[b,:] = o1 @ Wout^T + bout (1024 -> 2). ----
  {
    const float v1 = o1[tid], v2 = o1[tid + 512];
    float a0 = v1 * Wout[tid]         + v2 * Wout[tid + 512];
    float a1 = v1 * Wout[INNER + tid] + v2 * Wout[INNER + tid + 512];
#pragma unroll
    for (int m = 16; m >= 1; m >>= 1) {
      a0 += __shfl_xor(a0, m, 32);
      a1 += __shfl_xor(a1, m, 32);
    }
    if (lane == 0) { atomicAdd(&red[0], a0); atomicAdd(&red[1], a1); }
  }
  __syncthreads();
  if (tid == 0) {
    out[b * 2 + 0] = red[0] + bout[0];
    out[b * 2 + 1] = red[1] + bout[1];
  }
}

extern "C" void kernel_launch(void* const* d_in, const int* in_sizes, int n_in,
                              void* d_out, int out_size, void* d_ws, size_t ws_size,
                              hipStream_t stream) {
  (void)in_sizes; (void)n_in; (void)out_size; (void)d_ws; (void)ws_size;
  const float* drug1 = (const float*)d_in[0];
  const float* drug2 = (const float*)d_in[1];
  const float* ln_w  = (const float*)d_in[2];
  const float* ln_b  = (const float*)d_in[3];
  const float* Wd    = (const float*)d_in[4];
  const float* Wout  = (const float*)d_in[5];
  const float* bout  = (const float*)d_in[6];
  float* out = (float*)d_out;

  coattn_fused_kernel<<<dim3(256), dim3(THREADS), SMEM_BYTES, stream>>>(
      drug1, drug2, ln_w, ln_b, Wd, Wout, bout, out);
}
